// InteractionHead_59708635349526
// MI455X (gfx1250) — matrix operand order repaired and checked by reading, not compiled, not run on hardware
//
#include <hip/hip_runtime.h>
#include <hip/hip_bf16.h>
#include <stdint.h>
#include <stddef.h>

// ---------------------------------------------------------------------------
// Types for WMMA (CDNA5 / gfx1250, wave32)
// ---------------------------------------------------------------------------
typedef __attribute__((ext_vector_type(16))) __bf16 v16bf;
typedef __attribute__((ext_vector_type(8)))  __bf16 v8bf;
typedef __attribute__((ext_vector_type(8)))  float  v8f;

union FragBF { v16bf v; v8bf h8[2]; };

#define LDK 40  // LDS row stride in bf16 elements (80 bytes, multiple of 16B)

// ---------------------------------------------------------------------------
// Generic bf16 WMMA GEMM:  C = act(A[MxK] @ Bt[NxK]^T + bias)
//   A  : bf16 row-major MxK          (M % 128 == 0)
//   Bt : bf16 row-major NxK          (N % 64  == 0, K % 32 == 0)
//   Cf : optional f32 out, leading dim ldc
//   Cb : optional bf16 out, leading dim N
// Block: 256 threads = 8 waves; block tile 128x64; wave tile 32x32 (2x2 WMMA)
// Double-buffered LDS staging via GLOBAL_LOAD_ASYNC_TO_LDS_B128 (ASYNCcnt):
// tile k+1 streams into LDS while WMMAs consume tile k.
// ---------------------------------------------------------------------------
__global__ __launch_bounds__(256)
void gemm_bf16_wmma(const __bf16* __restrict__ A,
                    const __bf16* __restrict__ Bt,
                    const float*  __restrict__ bias,
                    float*  __restrict__ Cf,
                    __bf16* __restrict__ Cb,
                    int M, int N, int K, int ldc, int relu)
{
  __shared__ __bf16 sA[2][128 * LDK];
  __shared__ __bf16 sB[2][64 * LDK];

  const int tid  = threadIdx.x;
  const int lane = tid & 31;
  const int wave = tid >> 5;
  const int wm   = wave & 3;   // 4 waves along M
  const int wn   = wave >> 2;  // 2 waves along N
  const int tilesN = N >> 6;
  const int bm = (blockIdx.x / tilesN) << 7;
  const int bn = (blockIdx.x % tilesN) << 6;
  const int r  = lane & 15;
  const int hi = lane >> 4;

  v8f acc[2][2] = {};

  // cooperative staging: thread -> (row = tid/4, 16B at col (tid%4)*8)
  const int lrow = tid >> 2;
  const int lcol = (tid & 3) << 3;

  // per-thread LDS destination byte addresses (low 32 bits of generic ptr
  // == LDS offset per the flat-address aperture rules)
  unsigned la0[2], la1[2], lb0[2];
  la0[0] = (unsigned)(uintptr_t)&sA[0][lrow * LDK + lcol];
  la1[0] = (unsigned)(uintptr_t)&sA[0][(lrow + 64) * LDK + lcol];
  lb0[0] = (unsigned)(uintptr_t)&sB[0][lrow * LDK + lcol];
  la0[1] = (unsigned)(uintptr_t)&sA[1][lrow * LDK + lcol];
  la1[1] = (unsigned)(uintptr_t)&sA[1][(lrow + 64) * LDK + lcol];
  lb0[1] = (unsigned)(uintptr_t)&sB[1][lrow * LDK + lcol];

  // per-thread global source rows
  const __bf16* gArow0 = A  + (size_t)(bm + lrow)      * K + lcol;
  const __bf16* gArow1 = A  + (size_t)(bm + lrow + 64) * K + lcol;
  const __bf16* gBrow  = Bt + (size_t)(bn + lrow)      * K + lcol;

  // issue one 128x32 A-tile + 64x32 B-tile as async global->LDS copies
  auto issue_tile = [&](int buf, int koff) {
    asm volatile(
      "global_load_async_to_lds_b128 %0, %3, off\n\t"
      "global_load_async_to_lds_b128 %1, %4, off\n\t"
      "global_load_async_to_lds_b128 %2, %5, off"
      :
      : "v"(la0[buf]), "v"(la1[buf]), "v"(lb0[buf]),
        "v"((unsigned long long)(uintptr_t)(gArow0 + koff)),
        "v"((unsigned long long)(uintptr_t)(gArow1 + koff)),
        "v"((unsigned long long)(uintptr_t)(gBrow  + koff))
      : "memory");
  };

  const int tilesK = K >> 5;

  issue_tile(0, 0);
  asm volatile("s_wait_asynccnt 0x0" ::: "memory");
  __syncthreads();

  for (int t = 0; t < tilesK; ++t) {
    const int cur = t & 1;

    if (t + 1 < tilesK) issue_tile(cur ^ 1, (t + 1) << 5);
    if (t + 2 < tilesK) {  // gfx1250 global_prefetch_b8, distance-2 tiles
      __builtin_prefetch(gArow0 + ((t + 2) << 5), 0, 1);
      __builtin_prefetch(gBrow  + ((t + 2) << 5), 0, 1);
    }

    const __bf16* cA = &sA[cur][0];
    const __bf16* cB = &sB[cur][0];

    // A fragment (16x32 bf16): lane<16 -> K {0..7, 16..23}; lane>=16 -> +8
    FragBF a0, a1, b0, b1;
    const int am = wm << 5;
    a0.h8[0] = *(const v8bf*)&cA[(am + r)      * LDK + hi * 8];
    a0.h8[1] = *(const v8bf*)&cA[(am + r)      * LDK + hi * 8 + 16];
    a1.h8[0] = *(const v8bf*)&cA[(am + 16 + r) * LDK + hi * 8];
    a1.h8[1] = *(const v8bf*)&cA[(am + 16 + r) * LDK + hi * 8 + 16];
    // B fragment (32x16 bf16): lane<16 -> K {0..7, 16..23} of column r; lane>=16 -> +8
    const int bcol = wn << 5;
    b0.h8[0] = *(const v8bf*)&cB[(bcol + r)      * LDK + hi * 8];
    b0.h8[1] = *(const v8bf*)&cB[(bcol + r)      * LDK + hi * 8 + 16];
    b1.h8[0] = *(const v8bf*)&cB[(bcol + 16 + r) * LDK + hi * 8];
    b1.h8[1] = *(const v8bf*)&cB[(bcol + 16 + r) * LDK + hi * 8 + 16];

    acc[0][0] = __builtin_amdgcn_wmma_f32_16x16x32_bf16(false, a0.v, false, b0.v, (short)0, acc[0][0], false, false);
    acc[0][1] = __builtin_amdgcn_wmma_f32_16x16x32_bf16(false, a0.v, false, b1.v, (short)0, acc[0][1], false, false);
    acc[1][0] = __builtin_amdgcn_wmma_f32_16x16x32_bf16(false, a1.v, false, b0.v, (short)0, acc[1][0], false, false);
    acc[1][1] = __builtin_amdgcn_wmma_f32_16x16x32_bf16(false, a1.v, false, b1.v, (short)0, acc[1][1], false, false);

    if (t + 1 < tilesK)
      asm volatile("s_wait_asynccnt 0x0" ::: "memory");  // next buffer landed
    __syncthreads();                                     // publish to workgroup
  }

  // C/D layout: VGPR v -> row base + hi*8 + v ; lane r -> col
  #pragma unroll
  for (int ti = 0; ti < 2; ++ti) {
    #pragma unroll
    for (int tj = 0; tj < 2; ++tj) {
      const int col = bn + (wn << 5) + tj * 16 + r;
      const float bv = bias ? bias[col] : 0.f;
      #pragma unroll
      for (int v = 0; v < 8; ++v) {
        const int row = bm + (wm << 5) + ti * 16 + hi * 8 + v;
        float val = acc[ti][tj][v] + bv;
        if (relu) val = fmaxf(val, 0.f);
        if (Cf) Cf[(size_t)row * ldc + col] = val;
        if (Cb) Cb[(size_t)row * N + col] = (__bf16)val;
      }
    }
  }
}

// ---------------------------------------------------------------------------
// Weight preparation kernels
// ---------------------------------------------------------------------------
// Wt[n*Kp + k] = (k < K) ? W[k*N + n] : 0   (transpose + bf16 + K padding)
__global__ void conv_w_t(const float* __restrict__ W, __bf16* __restrict__ Wt,
                         int K, int N, int Kp) {
  int idx = blockIdx.x * 256 + threadIdx.x;
  if (idx >= N * Kp) return;
  int n = idx / Kp, k = idx - n * Kp;
  Wt[idx] = (k < K) ? (__bf16)W[(size_t)k * N + n] : (__bf16)0.f;
}

// (16, D, 32) -> Wt[(k*32+s)*D + d] = W[(k*D + d)*32 + s]
__global__ void conv_w_kds(const float* __restrict__ W, __bf16* __restrict__ Wt, int D) {
  int idx = blockIdx.x * 256 + threadIdx.x;
  if (idx >= 512 * D) return;
  int n = idx / D, d = idx - n * D;
  int k = n >> 5, s = n & 31;
  Wt[idx] = (__bf16)W[((size_t)k * D + d) * 32 + s];
}

// out[d] = sum_k B[k*512 + d]   (fold f3b/g3b.sum(0))
__global__ void bias_sum16(const float* __restrict__ B, float* __restrict__ o) {
  int d = blockIdx.x * 256 + threadIdx.x;
  if (d >= 512) return;
  float s = 0.f;
  #pragma unroll
  for (int k = 0; k < 16; ++k) s += B[k * 512 + d];
  o[d] = s;
}

// pairwise_spatial (49152 x 36) -> bf16 (49152 x 64) zero-padded
__global__ void pad_ps_kernel(const float* __restrict__ ps, __bf16* __restrict__ o) {
  size_t idx = (size_t)blockIdx.x * 256 + threadIdx.x;
  if (idx >= (size_t)49152 * 64) return;
  int c = (int)(idx & 63);
  size_t r = idx >> 6;
  o[idx] = (c < 36) ? (__bf16)ps[r * 36 + c] : (__bf16)0.f;
}

__global__ void cvt_f32_bf16(const float* __restrict__ a, __bf16* __restrict__ o, size_t n) {
  size_t i = (size_t)blockIdx.x * 256 + threadIdx.x;
  if (i < n) o[i] = (__bf16)a[i];
}

// ---------------------------------------------------------------------------
// g[b,d] = mean over 32x32 of features[b,d,:,:]; one wave per (b,d) row
// ---------------------------------------------------------------------------
__global__ __launch_bounds__(256)
void feat_mean_kernel(const float* __restrict__ f, float* __restrict__ g) {
  int wid  = (int)(((size_t)blockIdx.x * blockDim.x + threadIdx.x) >> 5);
  int lane = threadIdx.x & 31;
  if (wid >= 48 * 2048) return;
  const float4* p4 = (const float4*)(f + (size_t)wid * 1024);
  float s = 0.f;
  #pragma unroll
  for (int it = 0; it < 8; ++it) {
    float4 v = p4[lane + it * 32];
    s += v.x + v.y + v.z + v.w;
  }
  #pragma unroll
  for (int o = 16; o; o >>= 1) s += __shfl_xor(s, o, 32);
  if (lane == 0) g[wid] = s * (1.f / 1024.f);
}

// ---------------------------------------------------------------------------
// Attention: one wave per (b, h, j); lane = i (softmax over i, axis=2).
//   logit_i = aw[h,0:64].u_i + aw[h,64:128].u_j + aw[h,128:192].pr_ij + ab[h]
//   q[d]    = sum_i w_i * u_i[d] * pr_ij[d]        (since sum_i w_i == 1)
//   agg[e]  = relu( q @ mw[h] + mb[h] )  -> aggrelu[(b*32+j), h*64+e] (bf16)
// ---------------------------------------------------------------------------
__global__ __launch_bounds__(256)
void attn_kernel(const float* __restrict__ u, const float* __restrict__ pr,
                 const float* __restrict__ aw, const float* __restrict__ ab,
                 const float* __restrict__ mw, const float* __restrict__ mb,
                 __bf16* __restrict__ agg)
{
  int wid  = (int)(((size_t)blockIdx.x * blockDim.x + threadIdx.x) >> 5);
  int lane = threadIdx.x & 31;
  int j = wid & 31;
  int h = (wid >> 5) & 7;
  int b = wid >> 8;
  int i = lane;

  const float* awh  = aw + h * 192;
  const float* ui   = u  + (size_t)(b * 32 + i) * 512 + h * 64;
  const float* uj   = u  + (size_t)(b * 32 + j) * 512 + h * 64;
  const float* prij = pr + (size_t)((b * 32 + i) * 32 + j) * 512 + h * 64;

  float logit = ab[h];
  #pragma unroll 8
  for (int d = 0; d < 64; ++d)
    logit += awh[d] * ui[d] + awh[64 + d] * uj[d] + awh[128 + d] * prij[d];

  float mx = logit;
  #pragma unroll
  for (int o = 16; o; o >>= 1) mx = fmaxf(mx, __shfl_xor(mx, o, 32));
  float e  = __expf(logit - mx);
  float se = e;
  #pragma unroll
  for (int o = 16; o; o >>= 1) se += __shfl_xor(se, o, 32);
  float w = e / se;

  float q0 = 0.f, q1 = 0.f;  // lane owns d = lane and d = lane + 32
  for (int ii = 0; ii < 32; ++ii) {
    float wi = __shfl(w, ii, 32);
    const float* uv = u  + (size_t)(b * 32 + ii) * 512 + h * 64;
    const float* pv = pr + (size_t)((b * 32 + ii) * 32 + j) * 512 + h * 64;
    q0 += wi * uv[lane]      * pv[lane];
    q1 += wi * uv[lane + 32] * pv[lane + 32];
  }

  float a0 = mb[h * 64 + lane];
  float a1 = mb[h * 64 + lane + 32];
  const float* mwh = mw + (size_t)h * 64 * 64;
  for (int d = 0; d < 32; ++d) {
    float v = __shfl(q0, d, 32);
    a0 += v * mwh[d * 64 + lane];
    a1 += v * mwh[d * 64 + lane + 32];
  }
  for (int d = 0; d < 32; ++d) {
    float v = __shfl(q1, d, 32);
    a0 += v * mwh[(32 + d) * 64 + lane];
    a1 += v * mwh[(32 + d) * 64 + lane + 32];
  }
  size_t orow = (size_t)(b * 32 + j) * 512 + h * 64;
  agg[orow + lane]      = (__bf16)fmaxf(a0, 0.f);
  agg[orow + lane + 32] = (__bf16)fmaxf(a1, 0.f);
}

// ---------------------------------------------------------------------------
// x = LayerNorm(unary + aggout) * gamma + beta; one block per row (256 cols)
// ---------------------------------------------------------------------------
__global__ __launch_bounds__(256)
void add_ln_kernel(const float* __restrict__ unary, const float* __restrict__ agg,
                   const float* __restrict__ gamma, const float* __restrict__ beta,
                   float* __restrict__ x)
{
  __shared__ float red[256];
  int row = blockIdx.x, t = threadIdx.x;
  float v = unary[(size_t)row * 256 + t] + agg[(size_t)row * 256 + t];
  red[t] = v; __syncthreads();
  for (int s = 128; s; s >>= 1) { if (t < s) red[t] += red[t + s]; __syncthreads(); }
  float mu = red[0] * (1.f / 256.f);
  __syncthreads();
  float dv = v - mu;
  red[t] = dv * dv; __syncthreads();
  for (int s = 128; s; s >>= 1) { if (t < s) red[t] += red[t + s]; __syncthreads(); }
  float var = red[0] * (1.f / 256.f);
  x[(size_t)row * 256 + t] = dv * rsqrtf(var + 1e-5f) * gamma[t] + beta[t];
}

// ---------------------------------------------------------------------------
// app[m] = [x[b,I[m]], x[b,J[m]]] (bf16);  sp[m] = s_bf16[b, I[m], J[m]]
// one block (256 threads) per m row
// ---------------------------------------------------------------------------
__global__ __launch_bounds__(256)
void build_app_sp(const float* __restrict__ x, const __bf16* __restrict__ s,
                  __bf16* __restrict__ app, __bf16* __restrict__ sp)
{
  int m = blockIdx.x, t = threadIdx.x;
  int b = m >> 9, mm = m & 511;
  int i = mm >> 5, j = mm & 31;
  const float* xi = x + (size_t)(b * 32 + i) * 256;
  const float* xj = x + (size_t)(b * 32 + j) * 256;
  app[(size_t)m * 512 + t]       = (__bf16)xi[t];
  app[(size_t)m * 512 + 256 + t] = (__bf16)xj[t];
  const __bf16* srow = s + (size_t)((b * 32 + i) * 32 + j) * 512;
  sp[(size_t)m * 512 + t]       = srow[t];
  sp[(size_t)m * 512 + 256 + t] = srow[256 + t];
}

// h = bf16(relu(t1 * t2))
__global__ void mul_relu_bf16(const float* __restrict__ a, const float* __restrict__ b,
                              __bf16* __restrict__ o, size_t n) {
  size_t i = (size_t)blockIdx.x * 256 + threadIdx.x;
  if (i < n) { float v = a[i] * b[i]; o[i] = (__bf16)fmaxf(v, 0.f); }
}

// hg = bf16(relu(ag[b, n] * t2g[m, n])), b = m/512
__global__ void hg_mul_kernel(const float* __restrict__ ag, const float* __restrict__ t,
                              __bf16* __restrict__ o) {
  size_t i = (size_t)blockIdx.x * 256 + threadIdx.x;
  if (i >= (size_t)24576 * 512) return;
  int n = (int)(i & 511);
  int b = (int)(i >> 18);
  o[i] = (__bf16)fmaxf(ag[b * 512 + n] * t[i], 0.f);
}

// a_g[b, k*32+s] = g[b,:] . g1w[k,:,s] + g1b[k,s]
__global__ void ag_kernel(const float* __restrict__ g, const float* __restrict__ g1w,
                          const float* __restrict__ g1b, float* __restrict__ ag) {
  int idx = blockIdx.x * 256 + threadIdx.x;
  if (idx >= 48 * 512) return;
  int b = idx >> 9, n = idx & 511;
  int k = n >> 5, s = n & 31;
  const float* gv = g + b * 2048;
  const float* wv = g1w + (size_t)k * 2048 * 32 + s;
  float acc = g1b[n];
  for (int d = 0; d < 2048; ++d) acc += gv[d] * wv[(size_t)d * 32];
  ag[idx] = acc;
}

// ---------------------------------------------------------------------------
// Host-side orchestration
// ---------------------------------------------------------------------------
static inline int cdiv_i(size_t a, size_t b) { return (int)((a + b - 1) / b); }

extern "C" void kernel_launch(void* const* d_in, const int* in_sizes, int n_in,
                              void* d_out, int out_size, void* d_ws, size_t ws_size,
                              hipStream_t stream)
{
  (void)in_sizes; (void)n_in; (void)out_size; (void)ws_size;
  const float* features = (const float*)d_in[0];   // (48,2048,32,32)
  const float* unary    = (const float*)d_in[1];   // (48,32,256)
  const float* ps       = (const float*)d_in[2];   // (48,32,32,36)
  const float* sw1 = (const float*)d_in[3];  const float* sb1 = (const float*)d_in[4];
  const float* sw2 = (const float*)d_in[5];  const float* sb2 = (const float*)d_in[6];
  const float* sw3 = (const float*)d_in[7];  const float* sb3 = (const float*)d_in[8];
  const float* uw  = (const float*)d_in[9];  const float* ub  = (const float*)d_in[10];
  const float* pw  = (const float*)d_in[11]; const float* pb  = (const float*)d_in[12];
  const float* aw  = (const float*)d_in[13]; const float* ab  = (const float*)d_in[14];
  const float* mw  = (const float*)d_in[15]; const float* mb  = (const float*)d_in[16];
  const float* gw  = (const float*)d_in[17]; const float* gb  = (const float*)d_in[18];
  const float* lng = (const float*)d_in[19]; const float* lnb = (const float*)d_in[20];
  const float* f1w = (const float*)d_in[21]; const float* f1b = (const float*)d_in[22];
  const float* f2w = (const float*)d_in[23]; const float* f2b = (const float*)d_in[24];
  const float* f3w = (const float*)d_in[25]; const float* f3b = (const float*)d_in[26];
  const float* g1w = (const float*)d_in[27]; const float* g1b = (const float*)d_in[28];
  const float* g2w = (const float*)d_in[29]; const float* g2b = (const float*)d_in[30];
  const float* g3w = (const float*)d_in[31]; const float* g3b = (const float*)d_in[32];

  unsigned char* base = (unsigned char*)d_ws;
  size_t off = 0;
  auto carve = [&](size_t bytes) -> void* {
    void* p = base + off;
    off += (bytes + 255) & ~(size_t)255;
    return p;
  };

  // bf16 weights (N x K, transposed)
  __bf16* w_sw1t = (__bf16*)carve((size_t)128 * 64  * 2);
  __bf16* w_sw2t = (__bf16*)carve((size_t)256 * 128 * 2);
  __bf16* w_sw3t = (__bf16*)carve((size_t)512 * 256 * 2);
  __bf16* w_uwt  = (__bf16*)carve((size_t)512 * 256 * 2);
  __bf16* w_pwt  = (__bf16*)carve((size_t)512 * 512 * 2);
  __bf16* w_gwt  = (__bf16*)carve((size_t)256 * 512 * 2);
  __bf16* w_f1t  = (__bf16*)carve((size_t)512 * 512 * 2);
  __bf16* w_f2t  = (__bf16*)carve((size_t)512 * 512 * 2);
  __bf16* w_f3t  = (__bf16*)carve((size_t)512 * 512 * 2);
  __bf16* w_g2t  = (__bf16*)carve((size_t)512 * 512 * 2);
  __bf16* w_g3t  = (__bf16*)carve((size_t)512 * 512 * 2);
  float*  f3bsum = (float*)carve(512 * 4);
  float*  g3bsum = (float*)carve(512 * 4);
  // activations / intermediates
  __bf16* ps_pad = (__bf16*)carve((size_t)49152 * 64 * 2);
  __bf16* s1_bf  = (__bf16*)carve((size_t)49152 * 128 * 2);
  __bf16* s2_bf  = (__bf16*)carve((size_t)49152 * 256 * 2);
  __bf16* s_bf   = (__bf16*)carve((size_t)49152 * 512 * 2);
  __bf16* un_bf  = (__bf16*)carve((size_t)1536 * 256 * 2);
  float*  u_f    = (float*)carve((size_t)1536 * 512 * 4);
  float*  pr_f   = (float*)carve((size_t)49152 * 512 * 4);
  __bf16* aggrel = (__bf16*)carve((size_t)1536 * 512 * 2);
  float*  aggout = (float*)carve((size_t)1536 * 256 * 4);
  float*  x_f    = (float*)carve((size_t)1536 * 256 * 4);
  __bf16* app_bf = (__bf16*)carve((size_t)24576 * 512 * 2);
  __bf16* sp_bf  = (__bf16*)carve((size_t)24576 * 512 * 2);
  float*  t1_f   = (float*)carve((size_t)24576 * 512 * 4);
  float*  t2_f   = (float*)carve((size_t)24576 * 512 * 4);
  __bf16* h_bf   = (__bf16*)carve((size_t)24576 * 512 * 2);
  float*  g_f    = (float*)carve((size_t)48 * 2048 * 4);
  float*  ag_f   = (float*)carve((size_t)48 * 512 * 4);
  float*  t2g_f  = t1_f;                 // reuse (t1 dead after h)
  __bf16* hg_bf  = (__bf16*)t2_f;        // reuse (t2 dead after h)

  // -- weight prep --------------------------------------------------------
  conv_w_t<<<cdiv_i((size_t)128 * 64, 256),  256, 0, stream>>>(sw1, w_sw1t, 36, 128, 64);
  conv_w_t<<<cdiv_i((size_t)256 * 128, 256), 256, 0, stream>>>(sw2, w_sw2t, 128, 256, 128);
  conv_w_t<<<cdiv_i((size_t)512 * 256, 256), 256, 0, stream>>>(sw3, w_sw3t, 256, 512, 256);
  conv_w_t<<<cdiv_i((size_t)512 * 256, 256), 256, 0, stream>>>(uw,  w_uwt,  256, 512, 256);
  conv_w_t<<<cdiv_i((size_t)512 * 512, 256), 256, 0, stream>>>(pw,  w_pwt,  512, 512, 512);
  conv_w_t<<<cdiv_i((size_t)256 * 512, 256), 256, 0, stream>>>(gw,  w_gwt,  512, 256, 512);
  conv_w_t<<<cdiv_i((size_t)512 * 512, 256), 256, 0, stream>>>(f3w, w_f3t,  512, 512, 512);
  conv_w_t<<<cdiv_i((size_t)512 * 512, 256), 256, 0, stream>>>(g3w, w_g3t,  512, 512, 512);
  conv_w_kds<<<cdiv_i((size_t)512 * 512, 256), 256, 0, stream>>>(f1w, w_f1t, 512);
  conv_w_kds<<<cdiv_i((size_t)512 * 512, 256), 256, 0, stream>>>(f2w, w_f2t, 512);
  conv_w_kds<<<cdiv_i((size_t)512 * 512, 256), 256, 0, stream>>>(g2w, w_g2t, 512);
  bias_sum16<<<2, 256, 0, stream>>>(f3b, f3bsum);
  bias_sum16<<<2, 256, 0, stream>>>(g3b, g3bsum);

  // -- input prep ---------------------------------------------------------
  pad_ps_kernel<<<cdiv_i((size_t)49152 * 64, 256), 256, 0, stream>>>(ps, ps_pad);
  cvt_f32_bf16<<<cdiv_i((size_t)1536 * 256, 256), 256, 0, stream>>>(unary, un_bf, (size_t)1536 * 256);
  feat_mean_kernel<<<12288, 256, 0, stream>>>(features, g_f);     // 48*2048 waves

  // -- GEMM helper --------------------------------------------------------
  auto GEMM = [&](const __bf16* A, const __bf16* Bt, const float* bias,
                  float* Cf, __bf16* Cb, int M, int N, int K, int ldc, int relu) {
    gemm_bf16_wmma<<<(M / 128) * (N / 64), 256, 0, stream>>>(A, Bt, bias, Cf, Cb, M, N, K, ldc, relu);
  };

  // -- spatial MLP s1->s2->s ---------------------------------------------
  GEMM(ps_pad, w_sw1t, sb1, nullptr, s1_bf, 49152, 128, 64, 0, 1);
  GEMM(s1_bf,  w_sw2t, sb2, nullptr, s2_bf, 49152, 256, 128, 0, 1);
  GEMM(s2_bf,  w_sw3t, sb3, nullptr, s_bf,  49152, 512, 256, 0, 1);

  // -- u = relu(unary @ uw + ub), pr = relu(s @ pw + pb) ------------------
  GEMM(un_bf, w_uwt, ub, u_f,  nullptr, 1536,  512, 256, 512, 1);
  GEMM(s_bf,  w_pwt, pb, pr_f, nullptr, 49152, 512, 512, 512, 1);

  // -- attention -> aggrelu (bf16) ---------------------------------------
  attn_kernel<<<1536, 256, 0, stream>>>(u_f, pr_f, aw, ab, mw, mb, aggrel);  // 48*8*32 waves

  // -- aggout = aggrelu @ gw + gb; x = LN(unary + aggout) -----------------
  GEMM(aggrel, w_gwt, gb, aggout, nullptr, 1536, 256, 512, 256, 0);
  add_ln_kernel<<<1536, 256, 0, stream>>>(unary, aggout, lng, lnb, x_f);

  // -- gather app / sp ----------------------------------------------------
  build_app_sp<<<24576, 256, 0, stream>>>(x_f, s_bf, app_bf, sp_bf);

  // -- f path: t1, t2, h = relu(t1*t2), f_att -> out[:, 0:512] ------------
  GEMM(app_bf, w_f1t, f1b, t1_f, nullptr, 24576, 512, 512, 512, 0);
  GEMM(sp_bf,  w_f2t, f2b, t2_f, nullptr, 24576, 512, 512, 512, 0);
  mul_relu_bf16<<<cdiv_i((size_t)24576 * 512, 256), 256, 0, stream>>>(t1_f, t2_f, h_bf, (size_t)24576 * 512);
  GEMM(h_bf, w_f3t, f3bsum, (float*)d_out, nullptr, 24576, 512, 512, 1024, 1);

  // -- g path: a_g, t2g, hg = relu(a_g*t2g), f_g -> out[:, 512:1024] ------
  ag_kernel<<<cdiv_i((size_t)48 * 512, 256), 256, 0, stream>>>(g_f, g1w, g1b, ag_f);
  GEMM(sp_bf, w_g2t, g2b, t2g_f, nullptr, 24576, 512, 512, 512, 0);
  hg_mul_kernel<<<cdiv_i((size_t)24576 * 512, 256), 256, 0, stream>>>(ag_f, t2g_f, hg_bf);
  GEMM(hg_bf, w_g3t, g3bsum, (float*)d_out + 512, nullptr, 24576, 512, 512, 1024, 1);
}
